// DenseExpert_29429115912203
// MI455X (gfx1250) — compile-verified, matrix operand order checked by reading, not run on hardware
//
#include <hip/hip_runtime.h>

// DenseExpert: out[b,u] = sum_e g[b,e] * (x[b] @ alpha[e])[u] + (g @ beta)[b,u]
// Reformulated as ONE GEMM:  A'(65536 x 1024) x W(1024 x 128), where
//   A'[b, e*128+i] = g[b,e] * x[b,i]   (built on-the-fly from LDS, never in HBM)
//   W [e*128+i, u] = alpha[e,i,u]
// f16 WMMA (v_wmma_f32_16x16x32_f16) with f32 accumulation; bias folded into
// accumulator init.  B=65536, E=8, I=U=128.

typedef __attribute__((ext_vector_type(16))) _Float16 v16h;
typedef __attribute__((ext_vector_type(8)))  _Float16 v8h;
typedef __attribute__((ext_vector_type(8)))  float    v8f;
typedef __attribute__((ext_vector_type(4)))  float    f4;

#define NE 8
#define NI 128
#define NU 128
#define BM 128      // rows per workgroup
#define LDST 136    // LDS row stride in halves: 16B-aligned rows, stride 68 dwords
                    // (== 4 mod 64 banks -> 16 consecutive rows hit 16 distinct banks)

__global__ __launch_bounds__(256) void dense_expert_kernel(
    const float* __restrict__ x,      // (B, NI)
    const float* __restrict__ gate,   // (B, NE)
    const float* __restrict__ alpha,  // (NE, NI, NU)
    const float* __restrict__ beta,   // (NE, NU)
    float* __restrict__ out)          // (B, NU)
{
    __shared__ _Float16 sA[BM * LDST];   // x tile (f16), [row][i]
    __shared__ _Float16 sB[NU * LDST];   // alpha[e] transposed (f16), [u][i]
    __shared__ float    sG[BM * NE];     // gate tile
    __shared__ float    sBeta[NE * NU];  // bias pool

    const int tid  = threadIdx.x;
    const int lane = tid & 31;
    const int wave = tid >> 5;
    const int b0   = blockIdx.x * BM;

    // ---- stage gates + beta ----
    for (int idx = tid; idx < BM * NE; idx += 256)
        sG[idx] = gate[(size_t)b0 * NE + idx];
    for (int idx = tid; idx < NE * NU; idx += 256)
        sBeta[idx] = beta[idx];

    // ---- stage x tile as f16 (float4-vectorized, coalesced) ----
    {
        const f4* xg = (const f4*)(x + (size_t)b0 * NI);
        for (int v = tid; v < BM * NI / 4; v += 256) {
            f4 d = xg[v];
            int r = v >> 5;            // NI/4 = 32 float4 per row
            int c = (v & 31) * 4;
            _Float16* p = &sA[r * LDST + c];
            p[0] = (_Float16)d.x; p[1] = (_Float16)d.y;
            p[2] = (_Float16)d.z; p[3] = (_Float16)d.w;
        }
    }
    __syncthreads();

    // ---- per-lane WMMA fragment geometry (ISA 7.12.2 layouts) ----
    const int mloc  = lane & 15;          // A row / B column within 16
    const int khalf = (lane >> 4) * 8;    // K offset for upper lane half
    const int arow  = wave * 16 + mloc;   // this lane's A row in the tile

    // ---- init accumulators with bias:  acc = (g @ beta) tile ----
    v8f acc[8];
    #pragma unroll
    for (int t = 0; t < 8; ++t) {
        #pragma unroll
        for (int j = 0; j < 8; ++j) {
            // C/D layout: VGPR j -> M = j + 8*(lane>>4), N = lane&15
            int mrow = wave * 16 + (lane >> 4) * 8 + j;
            int ucol = t * 16 + mloc;
            float bias = 0.f;
            #pragma unroll
            for (int e = 0; e < NE; ++e)
                bias += sG[mrow * NE + e] * sBeta[e * NU + ucol];
            acc[t][j] = bias;
        }
    }

    // ---- expert loop: stage alpha[e]^T into LDS, then 4 K-chunks of WMMA ----
    #pragma unroll 1
    for (int e = 0; e < NE; ++e) {
        __syncthreads();  // previous-iteration sB reads complete before overwrite
        {
            const f4* ag = (const f4*)(alpha + (size_t)e * NI * NU);
            for (int v = tid; v < NI * NU / 4; v += 256) {
                f4 d = ag[v];
                int i  = v >> 5;           // NU/4 = 32 float4 per i-row
                int u4 = (v & 31) * 4;
                sB[(u4 + 0) * LDST + i] = (_Float16)d.x;
                sB[(u4 + 1) * LDST + i] = (_Float16)d.y;
                sB[(u4 + 2) * LDST + i] = (_Float16)d.z;
                sB[(u4 + 3) * LDST + i] = (_Float16)d.w;
            }
            // prefetch next expert's weights into cache while we compute
            if (e + 1 < NE) {
                const char* nxt = (const char*)(alpha + (size_t)(e + 1) * NI * NU);
                __builtin_prefetch(nxt + tid * 256, 0, 1);
            }
        }
        __syncthreads();

        // per-row gate scalar for this expert, splatted to packed f16
        const _Float16 gh = (_Float16)sG[arow * NE + e];
        v8h gsplat;
        #pragma unroll
        for (int q = 0; q < 8; ++q) gsplat[q] = gh;

        #pragma unroll
        for (int kc = 0; kc < NI / 32; ++kc) {
            const int k0 = kc * 32 + khalf;
            // A fragment = gate-scaled x slice (two b128 LDS loads + pk_mul)
            v8h alo = *(const v8h*)&sA[arow * LDST + k0];
            v8h ahi = *(const v8h*)&sA[arow * LDST + k0 + 16];
            alo = alo * gsplat;
            ahi = ahi * gsplat;
            v16h afrag;
            #pragma unroll
            for (int q = 0; q < 8; ++q) { afrag[q] = alo[q]; afrag[q + 8] = ahi[q]; }

            #pragma unroll
            for (int t = 0; t < 8; ++t) {
                const int ucol = t * 16 + mloc;
                v8h blo = *(const v8h*)&sB[ucol * LDST + k0];
                v8h bhi = *(const v8h*)&sB[ucol * LDST + k0 + 16];
                v16h bfrag;
                #pragma unroll
                for (int q = 0; q < 8; ++q) { bfrag[q] = blo[q]; bfrag[q + 8] = bhi[q]; }

                acc[t] = __builtin_amdgcn_wmma_f32_16x16x32_f16(
                    /*neg_a=*/false, afrag, /*neg_b=*/false, bfrag,
                    /*c_mod=*/(short)0, acc[t],
                    /*reuse_a=*/false, /*reuse_b=*/false);
            }
        }
    }

    // ---- epilogue: scatter accumulators (C/D VGPR layout) ----
    #pragma unroll
    for (int t = 0; t < 8; ++t) {
        const int ucol = t * 16 + mloc;
        #pragma unroll
        for (int j = 0; j < 8; ++j) {
            int mrow = wave * 16 + (lane >> 4) * 8 + j;
            out[(size_t)(b0 + mrow) * NU + ucol] = acc[t][j];
        }
    }
}

extern "C" void kernel_launch(void* const* d_in, const int* in_sizes, int n_in,
                              void* d_out, int out_size, void* d_ws, size_t ws_size,
                              hipStream_t stream) {
    const float* x     = (const float*)d_in[0];  // (65536, 128)
    const float* gatep = (const float*)d_in[1];  // (65536, 8)
    const float* alpha = (const float*)d_in[2];  // (8, 128, 128)
    const float* beta  = (const float*)d_in[3];  // (8, 128)
    float* out = (float*)d_out;                  // (65536, 128)

    const int Bn = in_sizes[0] / NI;             // 65536
    dim3 grid(Bn / BM);                          // 512 workgroups
    dim3 block(256);                             // 8 wave32s
    dense_expert_kernel<<<grid, block, 0, stream>>>(x, gatep, alpha, beta, out);
}
